// AttentionCell_70360154243398
// MI455X (gfx1250) — matrix operand order repaired and checked by reading, not compiled, not run on hardware
//
#include <hip/hip_runtime.h>

// ---------------------------------------------------------------------------
// BitNet attention cell for MI455X (gfx1250, wave32, WMMA).
// All heavy math runs through v_wmma_f32_16x16x32_f16.
// V chunks staged to LDS with CDNA5 async copy (ASYNCcnt path).
// ---------------------------------------------------------------------------

typedef __attribute__((ext_vector_type(16))) _Float16 v16h;
typedef __attribute__((ext_vector_type(8)))  float    v8f;

#define DEV_INLINE __device__ __forceinline__

DEV_INLINE v8f wmma_16x16x32(v16h a, v16h b, v8f c) {
  // D = A(16x32 f16) * B(32x16 f16) + C(16x16 f32)
  return __builtin_amdgcn_wmma_f32_16x16x32_f16(false, a, false, b, (short)0, c,
                                                false, false);
}

// A-fragment (16x32, f16), row-major source, contiguous K layout:
// lane<16 -> M=lane, khalf=0 ; lane>=16 -> M=lane-16, khalf=8
// VGPR v holds K = 2*khalf + v*2 , +1
DEV_INLINE v16h load_a_frag(const _Float16* A, int lda, int row_base, int k_base) {
  int lane  = threadIdx.x & 31;
  int row   = row_base + (lane & 15);
  int khalf = (lane < 16) ? 0 : 8;
  const _Float16* p = A + (size_t)row * lda + k_base;
  v16h a;
#pragma unroll
  for (int v = 0; v < 8; ++v) {
    int ks = 2 * khalf + v * 2;
    a[2 * v]     = p[ks];
    a[2 * v + 1] = p[ks + 1];
  }
  return a;
}

// B-fragment (32x16) where the B operand is W^T and W is [N x K] row-major:
// lane<16 -> N=col_base+lane, K=0..15 packed ; lane>=16 -> same N, K=16..31
DEV_INLINE v16h load_b_frag(const _Float16* W, int ldw, int col_base, int k_base) {
  int lane = threadIdx.x & 31;
  int n    = col_base + (lane & 15);
  int kh   = (lane < 16) ? 0 : 16;
  const _Float16* p = W + (size_t)n * ldw + k_base + kh;
  return *(const v16h*)p;  // 16 contiguous f16 (32B aligned for our strides)
}

// B-fragment where the source V is [K x N] row-major (contraction over rows).
DEV_INLINE v16h load_b_frag_strided(const _Float16* V, int ldv, int key_base,
                                    int hd_base) {
  int lane = threadIdx.x & 31;
  int n    = hd_base + (lane & 15);
  int kh   = (lane < 16) ? 0 : 16;
  v16h b;
#pragma unroll
  for (int v = 0; v < 8; ++v) {
    int k0 = key_base + kh + 2 * v;
    b[2 * v]     = V[(size_t)k0 * ldv + n];
    b[2 * v + 1] = V[(size_t)(k0 + 1) * ldv + n];
  }
  return b;
}

// ---------------------------------------------------------------------------
// Deterministic |w| mean (two stage)
// ---------------------------------------------------------------------------
__global__ void abs_sum_partial(const float* __restrict__ w, int n,
                                float* __restrict__ partial) {
  float s = 0.f;
  for (int i = blockIdx.x * blockDim.x + threadIdx.x; i < n;
       i += gridDim.x * blockDim.x)
    s += fabsf(w[i]);
#pragma unroll
  for (int m = 16; m; m >>= 1) s += __shfl_xor(s, m);
  __shared__ float red[8];
  int lane = threadIdx.x & 31, wid = threadIdx.x >> 5;
  if (lane == 0) red[wid] = s;
  __syncthreads();
  if (threadIdx.x == 0) {
    float t = 0.f;
    for (int i = 0; i < (int)(blockDim.x >> 5); ++i) t += red[i];
    partial[blockIdx.x] = t;
  }
}

__global__ void final_sum(const float* __restrict__ partial, int n,
                          float* __restrict__ out) {
  float s = 0.f;
  for (int i = threadIdx.x; i < n; i += blockDim.x) s += partial[i];
#pragma unroll
  for (int m = 16; m; m >>= 1) s += __shfl_xor(s, m);
  __shared__ float red[8];
  int lane = threadIdx.x & 31, wid = threadIdx.x >> 5;
  if (lane == 0) red[wid] = s;
  __syncthreads();
  if (threadIdx.x == 0) {
    float t = 0.f;
    for (int i = 0; i < (int)(blockDim.x >> 5); ++i) t += red[i];
    out[0] = t;
  }
}

// ---------------------------------------------------------------------------
// Ternary quantize (g != null) or plain f16 cast (g == null).
// Folds the RMSNorm gain g into the weight columns.
// ---------------------------------------------------------------------------
__global__ void quant_ternary(const float* __restrict__ w,
                              const float* __restrict__ g,
                              const float* __restrict__ absSum, float invCount,
                              _Float16* __restrict__ out, int total, int K) {
  int i = blockIdx.x * blockDim.x + threadIdx.x;
  if (i >= total) return;
  float v = w[i];
  if (g != nullptr) {
    float th = absSum[0] * invCount;
    float t  = (fabsf(v) > th) ? ((v > 0.f) ? 1.f : -1.f) : 0.f;
    v        = t * g[i % K];
  }
  out[i] = (_Float16)v;
}

// ---------------------------------------------------------------------------
// Row RMSNorm -> f16 (optionally also emit plain f16 cast of input).
// One 256-thread block per row.
// ---------------------------------------------------------------------------
template <typename Tin>
__global__ void rmsnorm_rows(const Tin* __restrict__ x, _Float16* __restrict__ xn,
                             _Float16* __restrict__ xc, int Dd) {
  int row = blockIdx.x;
  const Tin* xr = x + (size_t)row * Dd;
  float ss = 0.f;
  for (int i = threadIdx.x; i < Dd; i += blockDim.x) {
    float v = (float)xr[i];
    ss += v * v;
  }
#pragma unroll
  for (int m = 16; m; m >>= 1) ss += __shfl_xor(ss, m);
  __shared__ float red[8];
  __shared__ float stot;
  int lane = threadIdx.x & 31, wid = threadIdx.x >> 5;
  if (lane == 0) red[wid] = ss;
  __syncthreads();
  if (threadIdx.x == 0) {
    float t = 0.f;
    for (int i = 0; i < (int)(blockDim.x >> 5); ++i) t += red[i];
    stot = t;
  }
  __syncthreads();
  float r = rsqrtf(stot / (float)Dd + 1.1920929e-07f);
  for (int i = threadIdx.x; i < Dd; i += blockDim.x) {
    float v = (float)xr[i];
    xn[(size_t)row * Dd + i] = (_Float16)(v * r);
    if (xc) xc[(size_t)row * Dd + i] = (_Float16)v;
  }
}

// ---------------------------------------------------------------------------
// WMMA GEMM: Y[M,N] = A16[M,K] @ W16[N,K]^T.  One wave per 16x64 tile.
// Software-pipelined K loop (register double-buffer) so the scheduler can
// overlap next-step loads with the current WMMA block instead of draining
// LOADcnt to zero before every matrix op.
// ---------------------------------------------------------------------------
__global__ void gemm_wmma(const _Float16* __restrict__ A,
                          const _Float16* __restrict__ W,
                          _Float16* __restrict__ outH, float* __restrict__ outF,
                          int M, int N, int K) {
  int wave   = blockIdx.x * (blockDim.x >> 5) + (threadIdx.x >> 5);
  int ntiles = N >> 6;
  int tiles  = (M >> 4) * ntiles;
  if (wave >= tiles) return;
  int mt = wave / ntiles, nt = wave % ntiles;
  int row0 = mt << 4, col0 = nt << 6;
  int lane = threadIdx.x & 31;

  v8f acc0 = {}, acc1 = {}, acc2 = {}, acc3 = {};

  v16h a  = load_a_frag(A, K, row0, 0);
  v16h b0 = load_b_frag(W, K, col0 +  0, 0);
  v16h b1 = load_b_frag(W, K, col0 + 16, 0);
  v16h b2 = load_b_frag(W, K, col0 + 32, 0);
  v16h b3 = load_b_frag(W, K, col0 + 48, 0);

  for (int k = 32; k < K; k += 32) {
    // issue next-step loads first: they are independent of the WMMAs below
    v16h an  = load_a_frag(A, K, row0, k);
    v16h bn0 = load_b_frag(W, K, col0 +  0, k);
    v16h bn1 = load_b_frag(W, K, col0 + 16, k);
    v16h bn2 = load_b_frag(W, K, col0 + 32, k);
    v16h bn3 = load_b_frag(W, K, col0 + 48, k);
    __builtin_prefetch(A + (size_t)(row0 + (lane & 15)) * K + k + 96, 0, 3);
    __builtin_prefetch(W + (size_t)(col0 + (lane & 15)) * K + k + 96, 0, 3);
    acc0 = wmma_16x16x32(a, b0, acc0);
    acc1 = wmma_16x16x32(a, b1, acc1);
    acc2 = wmma_16x16x32(a, b2, acc2);
    acc3 = wmma_16x16x32(a, b3, acc3);
    a = an; b0 = bn0; b1 = bn1; b2 = bn2; b3 = bn3;
  }
  acc0 = wmma_16x16x32(a, b0, acc0);
  acc1 = wmma_16x16x32(a, b1, acc1);
  acc2 = wmma_16x16x32(a, b2, acc2);
  acc3 = wmma_16x16x32(a, b3, acc3);

  int cn   = lane & 15;
  int rofs = (lane < 16) ? 0 : 8;
  v8f acc[4] = {acc0, acc1, acc2, acc3};
#pragma unroll
  for (int t = 0; t < 4; ++t) {
#pragma unroll
    for (int r = 0; r < 8; ++r) {
      int row = row0 + r + rofs;
      int col = col0 + t * 16 + cn;
      if (outF) outF[(size_t)row * N + col] = acc[t][r];
      else      outH[(size_t)row * N + col] = (_Float16)acc[t][r];
    }
  }
}

// ---------------------------------------------------------------------------
// Flash attention, GQA, causal.  One wave per (b, h, 16-query tile).
// Q: [B*T, 1024] f16 ; K,V: [B*T, 256] f16 ; O: [B*T, 1024] f16
// V chunks (32 keys x 64 hd) are staged to LDS with CDNA5 async global->LDS
// copy, issued at the top of the chunk so the DMA overlaps S-WMMA + softmax.
// ---------------------------------------------------------------------------
__global__ void flash_attn(const _Float16* __restrict__ Q,
                           const _Float16* __restrict__ Kb,
                           const _Float16* __restrict__ Vb,
                           _Float16* __restrict__ O,
                           const float* __restrict__ qk_gain) {
  constexpr int T = 2048, D = 1024, DKV = 256, H = 16, HKV = 4;
  int wid  = threadIdx.x >> 5;
  int wave = blockIdx.x * (blockDim.x >> 5) + wid;
  int nqt  = T >> 4;
  int qt   = wave % nqt;
  int h    = (wave / nqt) % H;
  int b    = wave / (nqt * H);
  int kvh  = h / (H / HKV);
  int lane = threadIdx.x & 31;
  int cn   = lane & 15;
  int rofs = (lane < 16) ? 0 : 8;
  int qbase = qt << 4;

  const _Float16* Aq = Q  + (size_t)b * T * D   + h * 64;
  const _Float16* Bk = Kb + (size_t)b * T * DKV + kvh * 64;
  const _Float16* Bv = Vb + (size_t)b * T * DKV + kvh * 64;

  float scale = qk_gain[0] * 0.125f;  // 1/sqrt(64)

  v16h qa0 = load_a_frag(Aq, D, qbase, 0);
  v16h qa1 = load_a_frag(Aq, D, qbase, 32);

  v8f acc[4] = {};
  float m_run[8], l_run[8];
#pragma unroll
  for (int r = 0; r < 8; ++r) { m_run[r] = -3.0e38f; l_run[r] = 0.f; }

  __shared__ _Float16 plds_all[4][16 * 32];   // P tile (C-layout -> A-layout)
  __shared__ _Float16 vlds_all[4][32 * 64];   // async-staged V chunk
  _Float16* plds = plds_all[wid];
  _Float16* vlds = vlds_all[wid];

  // generic shared pointers truncate to the LDS byte offset (LDS aperture:
  // low 32 bits of the address); async copy wants that offset in a VGPR.
  uint32_t vlds_base = (uint32_t)(size_t)vlds;
  uint64_t vg_base   = (uint64_t)(size_t)Bv;

  int nchunk = (qbase + 16 + 31) >> 5;
  for (int c = 0; c < nchunk; ++c) {
    int kc = c << 5;

    // ---- async stage V chunk [kc..kc+31] x [0..63] f16 into LDS ----------
    // 4 KB chunk, 16 B per lane per instruction, 8 instructions per wave.
    // key = j*4 + lane/8 ; byte-in-row = (lane%8)*16 ; row stride = 512 B.
#pragma unroll
    for (int j = 0; j < 8; ++j) {
      int key  = j * 4 + (lane >> 3);
      int colb = (lane & 7) * 16;
      uint32_t voff = (uint32_t)((kc + key) * (DKV * 2) + colb);
      uint32_t ldst = vlds_base + (uint32_t)(key * 128 + colb);
      asm volatile("global_load_async_to_lds_b128 %0, %1, %2"
                   :: "v"(ldst), "v"(voff), "s"(vg_base)
                   : "memory");
    }

    // ---- S = Q K^T (two 16x16 key tiles, hd contraction split 2x32) ------
    v16h bk00 = load_b_frag(Bk, DKV, kc +  0, 0);
    v16h bk01 = load_b_frag(Bk, DKV, kc +  0, 32);
    v16h bk10 = load_b_frag(Bk, DKV, kc + 16, 0);
    v16h bk11 = load_b_frag(Bk, DKV, kc + 16, 32);
    v8f s0 = {}, s1 = {};
    s0 = wmma_16x16x32(qa0, bk00, s0);
    s0 = wmma_16x16x32(qa1, bk01, s0);
    s1 = wmma_16x16x32(qa0, bk10, s1);
    s1 = wmma_16x16x32(qa1, bk11, s1);

    // ---- online softmax in C-layout (row m = r+rofs, 16-lane half) -------
    float p0[8], p1[8];
#pragma unroll
    for (int r = 0; r < 8; ++r) {
      int qrow = qbase + r + rofs;
      float v0 = (kc + cn      <= qrow) ? s0[r] * scale : -3.0e38f;
      float v1 = (kc + 16 + cn <= qrow) ? s1[r] * scale : -3.0e38f;
      float cm = fmaxf(v0, v1);
      cm = fmaxf(cm, __shfl_xor(cm, 1));
      cm = fmaxf(cm, __shfl_xor(cm, 2));
      cm = fmaxf(cm, __shfl_xor(cm, 4));
      cm = fmaxf(cm, __shfl_xor(cm, 8));
      float mn = fmaxf(m_run[r], cm);
      float f  = __expf(m_run[r] - mn);
      float e0 = __expf(v0 - mn);
      float e1 = __expf(v1 - mn);
      float ps = e0 + e1;
      ps += __shfl_xor(ps, 1);
      ps += __shfl_xor(ps, 2);
      ps += __shfl_xor(ps, 4);
      ps += __shfl_xor(ps, 8);
      l_run[r] = l_run[r] * f + ps;
      m_run[r] = mn;
      p0[r] = e0;
      p1[r] = e1;
      acc[0][r] = acc[0][r] * f;
      acc[1][r] = acc[1][r] * f;
      acc[2][r] = acc[2][r] * f;
      acc[3][r] = acc[3][r] * f;
    }

    // ---- transpose P from C-layout to A-layout through LDS ---------------
#pragma unroll
    for (int r = 0; r < 8; ++r) {
      plds[(r + rofs) * 32 + cn]      = (_Float16)p0[r];
      plds[(r + rofs) * 32 + 16 + cn] = (_Float16)p1[r];
    }
    asm volatile("s_wait_dscnt 0" ::: "memory");  // CDNA5 split DS counter
    v16h pa = load_a_frag(plds, 32, 0, 0);

    // ---- drain async copy, gather V B-fragments from LDS -----------------
    asm volatile("s_wait_asynccnt 0" ::: "memory");
    v16h vb0 = load_b_frag_strided(vlds, 64, 0, 0);
    v16h vb1 = load_b_frag_strided(vlds, 64, 0, 16);
    v16h vb2 = load_b_frag_strided(vlds, 64, 0, 32);
    v16h vb3 = load_b_frag_strided(vlds, 64, 0, 48);

    // ---- O += P V (contraction over 32 keys, 4 hd tiles) -----------------
    acc[0] = wmma_16x16x32(pa, vb0, acc[0]);
    acc[1] = wmma_16x16x32(pa, vb1, acc[1]);
    acc[2] = wmma_16x16x32(pa, vb2, acc[2]);
    acc[3] = wmma_16x16x32(pa, vb3, acc[3]);
  }

#pragma unroll
  for (int r = 0; r < 8; ++r) {
    float inv = 1.0f / l_run[r];
    int row = qbase + r + rofs;
    size_t base = ((size_t)b * T + row) * D + h * 64;
#pragma unroll
    for (int t = 0; t < 4; ++t)
      O[base + t * 16 + cn] = (_Float16)(acc[t][r] * inv);
  }
}

// ---------------------------------------------------------------------------
// Host side
// ---------------------------------------------------------------------------
extern "C" void kernel_launch(void* const* d_in, const int* in_sizes, int n_in,
                              void* d_out, int out_size, void* d_ws,
                              size_t ws_size, hipStream_t stream) {
  const float* x       = (const float*)d_in[0];
  const float* q_w     = (const float*)d_in[1];
  const float* q_g     = (const float*)d_in[2];
  const float* k_w     = (const float*)d_in[3];
  const float* k_g     = (const float*)d_in[4];
  const float* v_w     = (const float*)d_in[5];
  const float* o_w     = (const float*)d_in[6];
  const float* o_g     = (const float*)d_in[7];
  const float* qk_gain = (const float*)d_in[8];

  constexpr int Bb = 2, T = 2048, D = 1024, DKV = 256;
  constexpr int MT = Bb * T;  // 4096 rows

  char* w = (char*)d_ws;
  auto carve = [&](size_t bytes) {
    char* p = w;
    w += (bytes + 255) & ~(size_t)255;
    return p;
  };
  float*     sums  = (float*)carve(3 * sizeof(float));
  float*     part  = (float*)carve(3 * 512 * sizeof(float));
  _Float16*  qw16  = (_Float16*)carve((size_t)D * D * 2);
  _Float16*  kw16  = (_Float16*)carve((size_t)DKV * D * 2);
  _Float16*  vw16  = (_Float16*)carve((size_t)DKV * D * 2);
  _Float16*  ow16  = (_Float16*)carve((size_t)D * D * 2);
  _Float16*  xn16  = (_Float16*)carve((size_t)MT * D * 2);
  _Float16*  xc16  = (_Float16*)carve((size_t)MT * D * 2);
  _Float16*  Q16   = (_Float16*)carve((size_t)MT * D * 2);
  _Float16*  K16   = (_Float16*)carve((size_t)MT * DKV * 2);
  _Float16*  V16   = (_Float16*)carve((size_t)MT * DKV * 2);
  _Float16*  A16   = (_Float16*)carve((size_t)MT * D * 2);

  // ternary thresholds (deterministic two-stage reduction)
  abs_sum_partial<<<512, 256, 0, stream>>>(q_w, D * D, part + 0 * 512);
  abs_sum_partial<<<512, 256, 0, stream>>>(k_w, DKV * D, part + 1 * 512);
  abs_sum_partial<<<512, 256, 0, stream>>>(o_w, D * D, part + 2 * 512);
  final_sum<<<1, 256, 0, stream>>>(part + 0 * 512, 512, sums + 0);
  final_sum<<<1, 256, 0, stream>>>(part + 1 * 512, 512, sums + 1);
  final_sum<<<1, 256, 0, stream>>>(part + 2 * 512, 512, sums + 2);

  // quantize / convert weights (gain folded into columns)
  quant_ternary<<<(D * D + 255) / 256, 256, 0, stream>>>(
      q_w, q_g, sums + 0, 1.f / (float)(D * D), qw16, D * D, D);
  quant_ternary<<<(DKV * D + 255) / 256, 256, 0, stream>>>(
      k_w, k_g, sums + 1, 1.f / (float)(DKV * D), kw16, DKV * D, D);
  quant_ternary<<<(D * D + 255) / 256, 256, 0, stream>>>(
      o_w, o_g, sums + 2, 1.f / (float)(D * D), ow16, D * D, D);
  quant_ternary<<<(DKV * D + 255) / 256, 256, 0, stream>>>(
      v_w, nullptr, nullptr, 0.f, vw16, DKV * D, D);

  // RMSNorm(x) -> f16 and plain f16 cast of x
  rmsnorm_rows<float><<<MT, 256, 0, stream>>>(x, xn16, xc16, D);

  // projections
  {
    int tiles = (MT / 16) * (D / 64);
    gemm_wmma<<<(tiles + 3) / 4, 128, 0, stream>>>(xn16, qw16, Q16, nullptr, MT, D, D);
  }
  {
    int tiles = (MT / 16) * (DKV / 64);
    gemm_wmma<<<(tiles + 3) / 4, 128, 0, stream>>>(xn16, kw16, K16, nullptr, MT, DKV, D);
    gemm_wmma<<<(tiles + 3) / 4, 128, 0, stream>>>(xc16, vw16, V16, nullptr, MT, DKV, D);
  }

  // causal GQA flash attention
  {
    int waves = Bb * 16 * (T / 16);  // 4096 waves
    flash_attn<<<waves / 4, 128, 0, stream>>>(Q16, K16, V16, A16, qk_gain);
  }

  // output bitlinear: RMSNorm(attn) @ ternary(o_w)^T -> fp32 d_out
  rmsnorm_rows<_Float16><<<MT, 256, 0, stream>>>(A16, xn16, (_Float16*)nullptr, D);
  {
    int tiles = (MT / 16) * (D / 64);
    gemm_wmma<<<(tiles + 3) / 4, 128, 0, stream>>>(xn16, ow16, nullptr, (float*)d_out,
                                                   MT, D, D);
  }
}
